// Quant3Linear_normal_eora_dummy_32762010534270
// MI455X (gfx1250) — compile-verified, matrix operand order checked by reading, not run on hardware
//
#include <hip/hip_runtime.h>
#include <hip/hip_bf16.h>

#define IN_F  16384
#define OUT_F 16384
#define RANK  64
#define KSPLIT 8
#define KCHUNK (IN_F / KSPLIT)      // 2048
#define GRP_PER_BLK (KCHUNK / 32)   // 64

#define KA_BLOCKS 32
#define KA_CHUNK  (IN_F / KA_BLOCKS) // 512 K-values per block
#define BSTRIDE   520                // padded LDS row stride (halves), 1040 B

typedef __attribute__((ext_vector_type(16))) _Float16 v16h;
typedef __attribute__((ext_vector_type(8)))  _Float16 v8h;
typedef __attribute__((ext_vector_type(8)))  float    v8f;

// ---------------------------------------------------------------------------
// Zero the atomic-accumulator regions of the workspace: A[OUT_F], t[RANK], S.
// ---------------------------------------------------------------------------
__global__ void zero_kernel(float* __restrict__ p, int n) {
  int i = blockIdx.x * blockDim.x + threadIdx.x;
  if (i < n) p[i] = 0.0f;
}

// ---------------------------------------------------------------------------
// Kernel A: t[r] += sum_i fp16(x[i]) * down[i,r]  via v_wmma_f32_16x16x32_f16
//           S    += sum_i fp16(x[i])
// 32 blocks x 256 threads; block stages its 512xRANK down-chunk into LDS
// *transposed* so every WMMA operand is two ds_load_b128 + a shuffle.
// A-tile is 16x32 f16 with only row M=0 (the x vector) non-zero.
// ---------------------------------------------------------------------------
__global__ __launch_bounds__(256) void lora_down_kernel(
    const float* __restrict__ x, const _Float16* __restrict__ down,
    float* __restrict__ ws_t, float* __restrict__ ws_S) {
  __shared__ __align__(16) _Float16 sBt[RANK * BSTRIDE]; // ~65 KB
  __shared__ __align__(16) _Float16 sxh[KA_CHUNK];       // 1 KB
  __shared__ float red[256];

  const int tid = threadIdx.x;
  const int k0  = blockIdx.x * KA_CHUNK;

  // stage fp16-rounded x chunk + S partial
  float s = 0.0f;
  for (int i = tid; i < KA_CHUNK; i += 256) {
    const _Float16 h = (_Float16)x[k0 + i];
    sxh[i] = h;
    s += (float)h;
  }
  red[tid] = s;

  // stage down chunk transposed: sBt[r*BSTRIDE + k] = down[k0+k, r]
  for (int idx = tid; idx < KA_CHUNK * RANK; idx += 256) {
    const int k = idx >> 6;      // idx / RANK
    const int r = idx & 63;      // idx % RANK
    sBt[r * BSTRIDE + k] = down[(size_t)(k0 + k) * RANK + r];
  }
  __syncthreads();

  const int wave = tid >> 5;
  const int lane = tid & 31;
  const int n    = lane & 15;
  // CDNA5 16-bit operand K packing: lanes 0-15 -> K {0..7, 16..23},
  //                                 lanes 16-31 -> K {8..15, 24..31}
  const int loOff = (lane < 16) ? 0 : 8;
  const int hiOff = (lane < 16) ? 16 : 24;
  const bool rowlane = (lane == 0) || (lane == 16); // holds A row M=0
  const v16h vzero = {};

  v8f c0{}, c1{}, c2{}, c3{};
#pragma unroll
  for (int it = 0; it < KA_CHUNK / (8 * 32); ++it) {   // 2 iters of K=32
    const int kl = wave * (KA_CHUNK / 8) + it * 32;

    const v8h alo = *reinterpret_cast<const v8h*>(&sxh[kl + loOff]);
    const v8h ahi = *reinterpret_cast<const v8h*>(&sxh[kl + hiOff]);
    v16h a = __builtin_shufflevector(alo, ahi, 0, 1, 2, 3, 4, 5, 6, 7,
                                     8, 9, 10, 11, 12, 13, 14, 15);
    if (!rowlane) a = vzero;                           // rows M=1..15 are zero

#define BLOAD(r0)                                                           \
  __builtin_shufflevector(                                                  \
      *reinterpret_cast<const v8h*>(&sBt[(r0 + n) * BSTRIDE + kl + loOff]), \
      *reinterpret_cast<const v8h*>(&sBt[(r0 + n) * BSTRIDE + kl + hiOff]), \
      0, 1, 2, 3, 4, 5, 6, 7, 8, 9, 10, 11, 12, 13, 14, 15)

    const v16h b0 = BLOAD(0);
    const v16h b1 = BLOAD(16);
    const v16h b2 = BLOAD(32);
    const v16h b3 = BLOAD(48);
#undef BLOAD

    c0 = __builtin_amdgcn_wmma_f32_16x16x32_f16(false, a, false, b0, (short)0, c0, false, false);
    c1 = __builtin_amdgcn_wmma_f32_16x16x32_f16(false, a, false, b1, (short)0, c1, false, false);
    c2 = __builtin_amdgcn_wmma_f32_16x16x32_f16(false, a, false, b2, (short)0, c2, false, false);
    c3 = __builtin_amdgcn_wmma_f32_16x16x32_f16(false, a, false, b3, (short)0, c3, false, false);
  }

  // C/D layout: row M=0 = element 0 of the v8f on lanes 0..15 (N = lane)
  if (lane < 16) {
    atomicAdd(&ws_t[ 0 + lane], c0[0]);
    atomicAdd(&ws_t[16 + lane], c1[0]);
    atomicAdd(&ws_t[32 + lane], c2[0]);
    atomicAdd(&ws_t[48 + lane], c3[0]);
  }

  __syncthreads();
  for (int off = 128; off > 0; off >>= 1) {
    if (tid < off) red[tid] += red[tid + off];
    __syncthreads();
  }
  if (tid == 0) atomicAdd(ws_S, red[0]);
}

// ---------------------------------------------------------------------------
// Kernel B: split-K GPTQ-3bit GEMV.  acc[o] = sum_i fp16(x[i]) * (8 + q3[i,o])
// Magic-number dequant: float bits (0x41000000 | q3<<20) == 8 + q3.
// The 8*S bias is removed in the finalize kernel.
// ---------------------------------------------------------------------------
__global__ __launch_bounds__(256) void main_gemv_kernel(
    const float* __restrict__ x, const unsigned* __restrict__ qw,
    float* __restrict__ wsA) {
  __shared__ __align__(16) float sx[KCHUNK];

  const int o     = blockIdx.x * 256 + threadIdx.x;
  const int kbase = blockIdx.y * KCHUNK;

  // stage fp16-rounded x chunk into LDS (broadcast-read later)
  for (int i = threadIdx.x; i < KCHUNK; i += 256)
    sx[i] = (float)(_Float16)x[kbase + i];
  __syncthreads();

  const int g0 = kbase >> 5;                   // first 32-value group
  const unsigned* qp = qw + (size_t)(3 * g0) * OUT_F + o;

  float acc = 0.0f;
  for (int g = 0; g < GRP_PER_BLK; ++g) {
    const unsigned w0 = qp[0];
    const unsigned w1 = qp[(size_t)OUT_F];
    const unsigned w2 = qp[(size_t)2 * OUT_F];
    qp += (size_t)3 * OUT_F;
    if (g + 4 < GRP_PER_BLK)
      __builtin_prefetch(qp + (size_t)9 * OUT_F, 0, 1);   // global_prefetch_b8

    const float4* xv = reinterpret_cast<const float4*>(&sx[g * 32]);
    float xs[32];
#pragma unroll
    for (int q = 0; q < 8; ++q) {
      const float4 v = xv[q];                  // ds_load_b128 broadcast
      xs[4 * q + 0] = v.x; xs[4 * q + 1] = v.y;
      xs[4 * q + 2] = v.z; xs[4 * q + 3] = v.w;
    }

    const unsigned wv[3] = {w0, w1, w2};
#pragma unroll
    for (int j = 0; j < 32; ++j) {
      const int b = 3 * j, wi = b >> 5, sft = b & 31;
      unsigned t;
      if (sft <= 29) {                         // field inside one word
        const unsigned w = wv[wi];
        t = (sft <= 20) ? (w << (20 - sft)) : (w >> (sft - 20));
      } else {                                 // spans word boundary (j=10, 21)
        t = ((wv[wi] >> sft) | (wv[wi + 1] << (32 - sft))) << 20;
      }
      const float f = __uint_as_float((t & 0x00700000u) | 0x41000000u); // 8+q3
      acc = fmaf(xs[j], f, acc);
    }
  }
  atomicAdd(&wsA[o], acc);
}

// ---------------------------------------------------------------------------
// Kernel C: y[o] = scale[o]*(A[o]-8S) - zero[o]*S + bias[o] + sum_r t[r]*up[r,o]
// ---------------------------------------------------------------------------
__global__ __launch_bounds__(256) void finalize_kernel(
    const float* __restrict__ wsA, const float* __restrict__ ws_t,
    const float* __restrict__ ws_S, const float* __restrict__ scales,
    const float* __restrict__ zeros, const float* __restrict__ bias,
    const _Float16* __restrict__ up, float* __restrict__ y) {
  __shared__ float ts[RANK];
  if (threadIdx.x < RANK) ts[threadIdx.x] = ws_t[threadIdx.x];
  __syncthreads();

  const int o = blockIdx.x * 256 + threadIdx.x;
  const float S = ws_S[0];
  const float A = wsA[o];

  float lora = 0.0f;
#pragma unroll 8
  for (int r = 0; r < RANK; ++r)
    lora = fmaf(ts[r], (float)up[(size_t)r * OUT_F + o], lora);

  const float sc = scales[o], z = zeros[o];
  y[o] = fmaf(sc, A - 8.0f * S, fmaf(-z, S, bias[o] + lora));
}

// ---------------------------------------------------------------------------
extern "C" void kernel_launch(void* const* d_in, const int* in_sizes, int n_in,
                              void* d_out, int out_size, void* d_ws, size_t ws_size,
                              hipStream_t stream) {
  const float*    x      = (const float*)d_in[0];
  const unsigned* qw     = (const unsigned*)d_in[1];
  const float*    scales = (const float*)d_in[2];
  const float*    zeros  = (const float*)d_in[3];
  const float*    bias   = (const float*)d_in[4];
  const _Float16* down   = (const _Float16*)d_in[5];
  const _Float16* up     = (const _Float16*)d_in[6];

  float* wsA  = (float*)d_ws;        // OUT_F accumulators (atomic split-K)
  float* ws_t = wsA + OUT_F;         // RANK lora intermediates (atomic)
  float* ws_S = ws_t + RANK;         // 1 scalar: sum of fp16(x) (atomic)

  const int nzero = OUT_F + RANK + 1;
  zero_kernel<<<(nzero + 255) / 256, 256, 0, stream>>>(wsA, nzero);
  lora_down_kernel<<<KA_BLOCKS, 256, 0, stream>>>(x, down, ws_t, ws_S);
  dim3 gridB(OUT_F / 256, KSPLIT);
  main_gemv_kernel<<<gridB, 256, 0, stream>>>(x, qw, wsA);
  finalize_kernel<<<OUT_F / 256, 256, 0, stream>>>(wsA, ws_t, ws_S, scales,
                                                   zeros, bias, up,
                                                   (float*)d_out);
}
